// MLP_module_5342939316926
// MI455X (gfx1250) — matrix operand order repaired and checked by reading, not compiled, not run on hardware
//
#include <hip/hip_runtime.h>

typedef __bf16 bf16_t;
typedef __attribute__((ext_vector_type(16))) __bf16 v16bf;
typedef __attribute__((ext_vector_type(8)))  __bf16 v8bf;
typedef __attribute__((ext_vector_type(4)))  __bf16 v4bf;
typedef __attribute__((ext_vector_type(8)))  float  v8f;
typedef __attribute__((ext_vector_type(4)))  float  v4f;

#define N_NODES   100000
#define IN_DIM    512
#define OUT_DIM   512
#define BLOCK_M   32
#define KSTEP     32
#define NSTEPS    (IN_DIM / KSTEP)   // 16
#define LN_EPS    1e-5f
#define NEG_SLOPE 0.2f

// Fused: out = LeakyReLU(LayerNorm(X @ W^T + b) * gamma + beta)
// WG = 256 threads = 8 waves. Each WG: 32 rows x 512 cols.
// wave w: row_group = w>>2 (16 rows), col_block = w&3 (128 cols = 8 wmma N-tiles).
// Pipeline: next K-tile is loaded to registers while current K-tile's WMMAs run.
__global__ __launch_bounds__(256)
void gemm_ln_lrelu(const float* __restrict__ X, const float* __restrict__ W,
                   const float* __restrict__ bvec, const float* __restrict__ gamma,
                   const float* __restrict__ beta, float* __restrict__ out)
{
    __shared__ __align__(16) bf16_t Bsh[OUT_DIM * KSTEP];  // [n][k], k contiguous
    __shared__ __align__(16) bf16_t Ash[BLOCK_M * KSTEP];  // [m][k], k contiguous
    __shared__ float Psum[BLOCK_M][4];
    __shared__ float Psq [BLOCK_M][4];

    const int tid       = threadIdx.x;
    const int wave      = tid >> 5;
    const int lane      = tid & 31;
    const int half      = lane >> 4;   // 0: lanes 0-15, 1: lanes 16-31
    const int m16       = lane & 15;
    const int row_group = wave >> 2;   // 0..1
    const int col_block = wave & 3;    // 0..3
    const int row0      = blockIdx.x * BLOCK_M;

    // Per-thread staging geometry. Linear bf16 tile index L = (it*256 + tid)*4,
    // with tile layout [n][k]: n = L>>5, k = L&31, and LDS offset == L.
    const int La = tid * 4;                 // A-tile linear index
    const int am = La >> 5, ak = La & 31;
    const float* Xrow = X + (size_t)(row0 + am) * IN_DIM + ak;

    v4f wreg[16];
    v4f xreg;

    v8f acc[8];
    #pragma unroll
    for (int t = 0; t < 8; ++t) acc[t] = v8f{};

    // ---- prologue: fetch first K-tile into registers ----
    #pragma unroll
    for (int it = 0; it < 16; ++it) {
        int L = (it * 256 + tid) * 4;
        int n = L >> 5, k = L & 31;
        wreg[it] = *(const v4f*)(W + (size_t)n * IN_DIM + k);
    }
    xreg = __builtin_nontemporal_load((const v4f*)Xrow);

    for (int s = 0; s < NSTEPS; ++s) {
        __syncthreads();   // previous tile fully consumed before overwrite

        // ---- convert staged registers -> bf16 LDS ----
        #pragma unroll
        for (int it = 0; it < 16; ++it) {
            int L = (it * 256 + tid) * 4;
            v4f w4 = wreg[it];
            v4bf p = { (bf16_t)w4.x, (bf16_t)w4.y, (bf16_t)w4.z, (bf16_t)w4.w };
            *(v4bf*)(&Bsh[L]) = p;
        }
        {
            v4bf p = { (bf16_t)xreg.x, (bf16_t)xreg.y, (bf16_t)xreg.z, (bf16_t)xreg.w };
            *(v4bf*)(&Ash[La]) = p;
        }
        __syncthreads();   // stores visible to all waves

        // ---- issue next tile's global loads now; waits deferred to next iter ----
        if (s + 1 < NSTEPS) {
            const int k0n = (s + 1) * KSTEP;
            #pragma unroll
            for (int it = 0; it < 16; ++it) {
                int L = (it * 256 + tid) * 4;
                int n = L >> 5, k = L & 31;
                wreg[it] = *(const v4f*)(W + (size_t)n * IN_DIM + k0n + k);
            }
            xreg = __builtin_nontemporal_load((const v4f*)(Xrow + k0n));
        }

        // ---- A fragment (16x32 bf16): lane<16 K={0..7,16..23}, lane>=16 K={8..15,24..31} ----
        const bf16_t* arow = &Ash[(row_group * 16 + m16) * KSTEP];
        v8bf alo = *(const v8bf*)(arow + half * 8);        // ds_load_b128
        v8bf ahi = *(const v8bf*)(arow + 16 + half * 8);   // ds_load_b128
        v16bf afrag = __builtin_shufflevector(alo, ahi,
            0,1,2,3,4,5,6,7,8,9,10,11,12,13,14,15);

        // ---- load all 8 B fragments, keep live so WMMAs issue back-to-back ----
        v16bf bfrag[8];
        #pragma unroll
        for (int t = 0; t < 8; ++t) {
            // B fragment (32x16 bf16): lane<16 K={0..7,16..23}, lane>=16 K={8..15,24..31}; N = lane&15
            const bf16_t* bcol = &Bsh[(col_block * 128 + t * 16 + m16) * KSTEP];
            v8bf b0 = *(const v8bf*)(bcol + half * 8);        // ds_load_b128
            v8bf b1 = *(const v8bf*)(bcol + 16 + half * 8);   // ds_load_b128
            bfrag[t] = __builtin_shufflevector(b0, b1,
                0,1,2,3,4,5,6,7,8,9,10,11,12,13,14,15);
        }
        #pragma unroll
        for (int t = 0; t < 8; ++t) {
            acc[t] = __builtin_amdgcn_wmma_f32_16x16x32_bf16(
                false, afrag, false, bfrag[t], (short)0, acc[t], false, false);
        }
    }

    // ---- bias (part of x_before, i.e. before LayerNorm stats) ----
    float gg[8], be[8];
    #pragma unroll
    for (int t = 0; t < 8; ++t) {
        int col = col_block * 128 + t * 16 + m16;
        float bias = bvec[col];
        gg[t] = gamma[col];
        be[t] = beta[col];
        #pragma unroll
        for (int r = 0; r < 8; ++r) acc[t][r] += bias;
    }

    // ---- per-row partial sums over this wave's 128 columns ----
    // acc[t][r] holds row M = r + 8*half (within 16-row group), col = t*16 + m16
    float psum[8], psq[8];
    #pragma unroll
    for (int r = 0; r < 8; ++r) { psum[r] = 0.f; psq[r] = 0.f; }
    #pragma unroll
    for (int t = 0; t < 8; ++t)
        #pragma unroll
        for (int r = 0; r < 8; ++r) { float v = acc[t][r]; psum[r] += v; psq[r] += v * v; }

    // reduce across the 16 lanes of each half (xor masks keep 'half' bit fixed)
    #pragma unroll
    for (int mask = 1; mask < 16; mask <<= 1) {
        #pragma unroll
        for (int r = 0; r < 8; ++r) {
            psum[r] += __shfl_xor(psum[r], mask, 32);
            psq[r]  += __shfl_xor(psq[r],  mask, 32);
        }
    }
    if (m16 == 0) {
        #pragma unroll
        for (int r = 0; r < 8; ++r) {
            int row = row_group * 16 + 8 * half + r;
            Psum[row][col_block] = psum[r];
            Psq [row][col_block] = psq[r];
        }
    }
    __syncthreads();

    // ---- finalize: combine 4 col-blocks, normalize, affine, LeakyReLU, store ----
    #pragma unroll
    for (int r = 0; r < 8; ++r) {
        int row = row_group * 16 + 8 * half + r;
        float s = Psum[row][0] + Psum[row][1] + Psum[row][2] + Psum[row][3];
        float q = Psq [row][0] + Psq [row][1] + Psq [row][2] + Psq [row][3];
        float mean = s * (1.0f / OUT_DIM);
        float var  = q * (1.0f / OUT_DIM) - mean * mean;
        float rstd = rsqrtf(var + LN_EPS);
        size_t obase = (size_t)(row0 + row) * OUT_DIM;
        #pragma unroll
        for (int t = 0; t < 8; ++t) {
            int col = col_block * 128 + t * 16 + m16;
            float v = (acc[t][r] - mean) * rstd * gg[t] + be[t];
            float o = (v >= 0.f) ? v : NEG_SLOPE * v;
            __builtin_nontemporal_store(o, out + obase + col);
        }
    }
}

extern "C" void kernel_launch(void* const* d_in, const int* in_sizes, int n_in,
                              void* d_out, int out_size, void* d_ws, size_t ws_size,
                              hipStream_t stream) {
    (void)in_sizes; (void)n_in; (void)out_size; (void)d_ws; (void)ws_size;
    // setup_inputs order: x, edge_attr, edge_index, batch, W, b, gamma, beta
    // (edge_attr / edge_index / batch are unused by the reference output)
    const float* X     = (const float*)d_in[0];
    const float* W     = (const float*)d_in[4];
    const float* bvec  = (const float*)d_in[5];
    const float* gam   = (const float*)d_in[6];
    const float* bet   = (const float*)d_in[7];
    float* out = (float*)d_out;

    dim3 grid(N_NODES / BLOCK_M);  // 100000/32 = 3125, exact
    dim3 block(256);
    gemm_ln_lrelu<<<grid, block, 0, stream>>>(X, W, bvec, gam, bet, out);
}
